// attention_4638564680174
// MI455X (gfx1250) — matrix operand order repaired and checked by reading, not compiled, not run on hardware
//
#include <hip/hip_runtime.h>

// ---------------------------------------------------------------------------
// Types for CDNA5 WMMA (wave32): v_wmma_f32_16x16x32_bf16
// ---------------------------------------------------------------------------
typedef __attribute__((ext_vector_type(16))) __bf16 v16bf;
typedef __attribute__((ext_vector_type(8)))  float  v8f;
typedef __attribute__((ext_vector_type(4)))  float  f32x4;
typedef __attribute__((ext_vector_type(4)))  unsigned int u32x4;

union Frag  { u32x4 q[2]; v16bf v; };
union Pack8 { u32x4 q; unsigned short s[8]; unsigned int w[4]; };

// cheap round-to-nearest (ties away): 2 VALU
__device__ __forceinline__ unsigned short f32_to_bf16(float f) {
  return (unsigned short)((__float_as_uint(f) + 0x8000u) >> 16);
}
// two floats -> packed bf16x2 dword: add/add/v_perm, ~1.5 VALU per element
__device__ __forceinline__ unsigned int pack_bf16x2(float lo, float hi) {
  const unsigned a = __float_as_uint(lo) + 0x8000u;
  const unsigned b = __float_as_uint(hi) + 0x8000u;
  return (b & 0xFFFF0000u) | (a >> 16);
}

__device__ __forceinline__ v8f wmma_bf16(const Frag& a, const Frag& b, v8f c) {
  return __builtin_amdgcn_wmma_f32_16x16x32_bf16(
      false, a.v, false, b.v, (short)0, c, false, false);
}

#define BATCH 4
#define SEQ   4096
#define DIM   1024
#define N3    3072
#define MTOT  (BATCH * SEQ)   // 16384

// ---------------------------------------------------------------------------
// Kernel 1: QKV projection  C[16384,3072] = X[16384,1024] * W[1024,3072] + b
// Block tile 128x128, 8 waves as 4(M) x 2(N), wave tile 32x64 (8 WMMA/step).
// Double-buffered LDS: one barrier per k-step, global fetch overlaps WMMA.
// ---------------------------------------------------------------------------
__global__ __launch_bounds__(256)
void qkv_gemm_kernel(const float* __restrict__ X, const float* __restrict__ W,
                     const float* __restrict__ bias,
                     unsigned short* __restrict__ Qo,
                     unsigned short* __restrict__ Ko,
                     unsigned short* __restrict__ Vo)
{
  __shared__ unsigned short As[2][128 * 32];   // [m][k] bf16, double buffered
  __shared__ unsigned short Bs[2][128 * 32];   // [n][k] bf16 (transposed)

  const int m0 = blockIdx.x * 128;
  const int n0 = blockIdx.y * 128;
  const int t  = threadIdx.x;
  const int lane = t & 31, wave = t >> 5;
  const int wm = wave & 3, wn = wave >> 2;
  const int half = lane >> 4, l16 = lane & 15;

  v8f acc[2][4] = {};

  // staging map: A: 2 thr/row, 16 k-cols each; B: 8 thr/k-row, 16 n-cols each
  const int ar = t >> 1;
  const int ac = (t & 1) * 16;
  const int bk = t >> 3;
  const int bn = (t & 7) * 16;

  const float* ag = X + (size_t)(m0 + ar) * DIM + ac;
  const float* bg = W + (size_t)bk * N3 + n0 + bn;

  f32x4 arg[4], brg[4];

  auto fetch = [&](int k0) {
    #pragma unroll
    for (int i = 0; i < 4; ++i) arg[i] = *(const f32x4*)(ag + k0 + i * 4);
    const float* bp = bg + (size_t)k0 * N3;
    #pragma unroll
    for (int i = 0; i < 4; ++i) brg[i] = *(const f32x4*)(bp + i * 4);
  };
  auto stage = [&](int buf) {
    const float* fa = (const float*)arg;
    Pack8 p0, p1;
    #pragma unroll
    for (int i = 0; i < 4; ++i) {
      p0.w[i] = pack_bf16x2(fa[2 * i],     fa[2 * i + 1]);
      p1.w[i] = pack_bf16x2(fa[8 + 2 * i], fa[8 + 2 * i + 1]);
    }
    u32x4* d = (u32x4*)&As[buf][ar * 32 + ac];
    d[0] = p0.q; d[1] = p1.q;
    const float* fb = (const float*)brg;
    #pragma unroll
    for (int i = 0; i < 16; ++i) Bs[buf][(bn + i) * 32 + bk] = f32_to_bf16(fb[i]);
  };
  auto compute = [&](int buf) {
    Frag a[2];
    #pragma unroll
    for (int i = 0; i < 2; ++i) {
      const unsigned short* ab = &As[buf][(wm * 32 + i * 16 + l16) * 32 + half * 8];
      a[i].q[0] = *(const u32x4*)ab;
      a[i].q[1] = *(const u32x4*)(ab + 16);
    }
    #pragma unroll
    for (int j = 0; j < 4; ++j) {
      Frag bf;
      const unsigned short* bb = &Bs[buf][(wn * 64 + j * 16 + l16) * 32 + half * 8];
      bf.q[0] = *(const u32x4*)bb;
      bf.q[1] = *(const u32x4*)(bb + 16);
      #pragma unroll
      for (int i = 0; i < 2; ++i) acc[i][j] = wmma_bf16(a[i], bf, acc[i][j]);
    }
  };

  fetch(0);
  stage(0);
  __syncthreads();
  int buf = 0;
  for (int k0 = 0; k0 < DIM; k0 += 32) {
    const bool more = (k0 + 32) < DIM;
    if (more) fetch(k0 + 32);
    compute(buf);
    if (more) stage(buf ^ 1);
    __syncthreads();
    buf ^= 1;
  }

  // route into Q / K / V (a 128-col block span sits in one region)
  const int region = n0 >> 10;
  unsigned short* dst = (region == 0) ? Qo : ((region == 1) ? Ko : Vo);
  const int ncol0 = n0 & 1023;
  #pragma unroll
  for (int i = 0; i < 2; ++i) {
    #pragma unroll
    for (int j = 0; j < 4; ++j) {
      const int col = wn * 64 + j * 16 + l16;
      const float bb = bias[n0 + col];
      #pragma unroll
      for (int r = 0; r < 8; ++r) {
        const int row = m0 + wm * 32 + i * 16 + half * 8 + r;
        dst[(size_t)row * DIM + ncol0 + col] = f32_to_bf16(acc[i][j][r] + bb);
      }
    }
  }
}

// ---------------------------------------------------------------------------
// Kernel 2: scores.  Block: 32 query rows of one batch, Q tile in LDS (64KB).
// Each wave streams 64-key tiles: 8 WMMAs per d-step (2 q-sub x 4 key-sub)
// with full A-fragment reuse.  P = exp(s/32) bf16 (unnormalized); row sums
// accumulated via global f32 atomics into a zeroed rowsum buffer.
// ---------------------------------------------------------------------------
__global__ __launch_bounds__(256)
void attn_scores_kernel(const unsigned short* __restrict__ Q,
                        const unsigned short* __restrict__ K,
                        unsigned short* __restrict__ P,
                        float* __restrict__ rowsum)
{
  __shared__ unsigned short Qs[32 * DIM];   // 64 KB
  const int bidx = blockIdx.y;
  const int q0   = blockIdx.x * 32;
  const size_t qkBase = (size_t)bidx * SEQ * DIM;
  const int t = threadIdx.x;

  { // stage Q tile (32x1024 bf16)
    const u32x4* src = (const u32x4*)(Q + qkBase + (size_t)q0 * DIM);
    u32x4* dl = (u32x4*)Qs;
    for (int i = t; i < 32 * DIM / 8; i += 256) dl[i] = src[i];
  }
  __syncthreads();

  const int lane = t & 31, wave = t >> 5, half = lane >> 4, l16 = lane & 15;
  const size_t pBase = (size_t)bidx * SEQ * SEQ;
  float* rsum = rowsum + (size_t)bidx * SEQ + q0;

  for (int kb = wave; kb < SEQ / 64; kb += 8) {
    const int key0 = kb * 64;
    const unsigned short* kr0 = K + qkBase + (size_t)(key0 + l16) * DIM + half * 8;
    v8f s[2][4] = {};
    for (int d0 = 0; d0 < DIM; d0 += 32) {
      Frag a[2], bf[4];
      const unsigned short* ab0 = &Qs[l16 * DIM + d0 + half * 8];
      a[0].q[0] = *(const u32x4*)ab0;
      a[0].q[1] = *(const u32x4*)(ab0 + 16);
      const unsigned short* ab1 = ab0 + 16 * DIM;
      a[1].q[0] = *(const u32x4*)ab1;
      a[1].q[1] = *(const u32x4*)(ab1 + 16);
      #pragma unroll
      for (int c = 0; c < 4; ++c) {
        const unsigned short* kp = kr0 + (size_t)(c * 16) * DIM + d0;
        bf[c].q[0] = *(const u32x4*)kp;
        bf[c].q[1] = *(const u32x4*)(kp + 16);
      }
      #pragma unroll
      for (int i = 0; i < 2; ++i)
        #pragma unroll
        for (int c = 0; c < 4; ++c)
          s[i][c] = wmma_bf16(a[i], bf[c], s[i][c]);
    }
    #pragma unroll
    for (int i = 0; i < 2; ++i) {
      #pragma unroll
      for (int r = 0; r < 8; ++r) {
        const size_t prow = pBase + (size_t)(q0 + i * 16 + half * 8 + r) * SEQ + key0;
        float red = 0.f;
        #pragma unroll
        for (int c = 0; c < 4; ++c) {
          const float e = __expf(s[i][c][r] * 0.03125f);  // scale = D^-0.5
          P[prow + c * 16 + l16] = f32_to_bf16(e);
          red += e;
        }
        red += __shfl_xor(red, 1, 32);                    // row-sum over 16 lanes
        red += __shfl_xor(red, 2, 32);
        red += __shfl_xor(red, 4, 32);
        red += __shfl_xor(red, 8, 32);
        if (l16 == 0) atomicAdd(&rsum[i * 16 + half * 8 + r], red);
      }
    }
  }
}

// ---------------------------------------------------------------------------
// Kernel 3: O = (P . V) / rowsum.  Per batch bf16 GEMM M=4096,N=1024,K=4096.
// Same 128x128 double-buffered tiling as kernel 1.
// ---------------------------------------------------------------------------
__global__ __launch_bounds__(256)
void pv_gemm_kernel(const unsigned short* __restrict__ P,
                    const unsigned short* __restrict__ V,
                    const float* __restrict__ rowsum,
                    float* __restrict__ O)
{
  __shared__ unsigned short As[2][128 * 32];
  __shared__ unsigned short Bs[2][128 * 32];

  const int b  = blockIdx.z;
  const int m0 = blockIdx.x * 128;
  const int n0 = blockIdx.y * 128;
  const size_t pBase = (size_t)b * SEQ * SEQ;
  const size_t vBase = (size_t)b * SEQ * DIM;

  const int t = threadIdx.x;
  const int lane = t & 31, wave = t >> 5;
  const int wm = wave & 3, wn = wave >> 2;
  const int half = lane >> 4, l16 = lane & 15;

  v8f acc[2][4] = {};

  const int ar = t >> 1;
  const int ac = (t & 1) * 16;
  const int bk = t >> 3;
  const int bn = (t & 7) * 16;

  const unsigned short* ag = P + pBase + (size_t)(m0 + ar) * SEQ + ac;
  const unsigned short* vg = V + vBase + (size_t)bk * DIM + n0 + bn;

  u32x4 aR[2];
  Pack8 bR[2];

  auto fetch = [&](int k0) {
    aR[0] = *(const u32x4*)(ag + k0);
    aR[1] = *(const u32x4*)(ag + k0 + 8);
    const unsigned short* vp = vg + (size_t)k0 * DIM;
    bR[0].q = *(const u32x4*)(vp);
    bR[1].q = *(const u32x4*)(vp + 8);
  };
  auto stage = [&](int buf) {
    u32x4* d = (u32x4*)&As[buf][ar * 32 + ac];
    d[0] = aR[0]; d[1] = aR[1];
    #pragma unroll
    for (int i = 0; i < 8; ++i) {
      Bs[buf][(bn + i) * 32 + bk]     = bR[0].s[i];
      Bs[buf][(bn + 8 + i) * 32 + bk] = bR[1].s[i];
    }
  };
  auto compute = [&](int buf) {
    Frag a[2];
    #pragma unroll
    for (int i = 0; i < 2; ++i) {
      const unsigned short* ab = &As[buf][(wm * 32 + i * 16 + l16) * 32 + half * 8];
      a[i].q[0] = *(const u32x4*)ab;
      a[i].q[1] = *(const u32x4*)(ab + 16);
    }
    #pragma unroll
    for (int j = 0; j < 4; ++j) {
      Frag bf;
      const unsigned short* bb = &Bs[buf][(wn * 64 + j * 16 + l16) * 32 + half * 8];
      bf.q[0] = *(const u32x4*)bb;
      bf.q[1] = *(const u32x4*)(bb + 16);
      #pragma unroll
      for (int i = 0; i < 2; ++i) acc[i][j] = wmma_bf16(a[i], bf, acc[i][j]);
    }
  };

  fetch(0);
  stage(0);
  __syncthreads();
  int buf = 0;
  for (int k0 = 0; k0 < SEQ; k0 += 32) {
    const bool more = (k0 + 32) < SEQ;
    if (more) fetch(k0 + 32);
    compute(buf);
    if (more) stage(buf ^ 1);
    __syncthreads();
    buf ^= 1;
  }

  #pragma unroll
  for (int i = 0; i < 2; ++i) {
    #pragma unroll
    for (int r = 0; r < 8; ++r) {
      const int row = m0 + wm * 32 + i * 16 + half * 8 + r;
      const float inv = 1.f / rowsum[(size_t)b * SEQ + row];
      #pragma unroll
      for (int j = 0; j < 4; ++j) {
        const int col = n0 + wn * 64 + j * 16 + l16;
        O[((size_t)b * SEQ + row) * DIM + col] = acc[i][j][r] * inv;
      }
    }
  }
}

// ---------------------------------------------------------------------------
// Launch
// ---------------------------------------------------------------------------
extern "C" void kernel_launch(void* const* d_in, const int* in_sizes, int n_in,
                              void* d_out, int out_size, void* d_ws, size_t ws_size,
                              hipStream_t stream) {
  const float* x    = (const float*)d_in[0];   // [4,4096,1024] f32
  const float* Wqkv = (const float*)d_in[1];   // [1024,3072]   f32
  const float* bqkv = (const float*)d_in[2];   // [3072]        f32
  float* out = (float*)d_out;                  // [4,4096,1024] f32

  // workspace layout (bf16 QKV, bf16 P, f32 rowsum)
  unsigned short* Qw = (unsigned short*)d_ws;
  unsigned short* Kw = Qw + (size_t)MTOT * DIM;
  unsigned short* Vw = Kw + (size_t)MTOT * DIM;
  unsigned short* Pw = Vw + (size_t)MTOT * DIM;
  float* rs = (float*)(Pw + (size_t)BATCH * SEQ * SEQ);

  dim3 g1(MTOT / 128, N3 / 128);        // 128 x 24
  qkv_gemm_kernel<<<g1, 256, 0, stream>>>(x, Wqkv, bqkv, Qw, Kw, Vw);

  hipMemsetAsync(rs, 0, (size_t)BATCH * SEQ * sizeof(float), stream);

  dim3 g2(SEQ / 32, BATCH);             // 128 x 4
  attn_scores_kernel<<<g2, 256, 0, stream>>>(Qw, Kw, Pw, rs);

  dim3 g3(SEQ / 128, DIM / 128, BATCH); // 32 x 8 x 4
  pv_gemm_kernel<<<g3, 256, 0, stream>>>(Pw, Vw, rs, out);
}
